// ROI_50964081935328
// MI455X (gfx1250) — matrix operand order repaired and checked by reading, not compiled, not run on hardware
//
#include <hip/hip_runtime.h>

// ---------------------------------------------------------------------------
// ROI head: fused dual-GEMM  X[131072x1024] @ [W_cls | W_tf] (1024 x 105) + bias
// MI455X plan: memory-bound (~592 MB -> ~25us @ 23.3 TB/s). Stream X once,
// bf16 WMMA w/ f32 accum. B pre-packed bf16 in ws; staged to LDS in two
// 112KB K-halves via global_load_async_to_lds_b128 (2 WGs/WGP occupancy).
// Each wave owns 32 rows (two 16-row M-tiles) so one B fragment feeds 2 WMMAs.
// ---------------------------------------------------------------------------

typedef __bf16 bf16_t;
typedef __attribute__((ext_vector_type(16))) __bf16 v16bf;
typedef __attribute__((ext_vector_type(8)))  float  v8f;
typedef __attribute__((ext_vector_type(4)))  float  f32x4;

#define N_ROWS   131072
#define D_K      1024
#define C_CLS    21
#define C_TF     84
#define C_TOT    105
#define NT       7                        // 7 N-tiles of 16 (105 padded to 112)
#define KT       32                       // 32 K-tiles of 32
#define KT_HALF  16
#define FRAG_BYTES  1024                  // one B fragment: 32 lanes x 16 bf16
#define BPACK_ELEMS (NT * KT * 32 * 16)   // 114688 bf16
#define BPACK_BYTES (BPACK_ELEMS * 2)     // 229376
#define CHUNK_BYTES (NT * KT_HALF * FRAG_BYTES)  // 114688 B = 112 KB LDS
#define CHUNK_F4    (CHUNK_BYTES / 16)    // 7168 16-byte chunks
#define BIAS_OFF    BPACK_BYTES           // fp32 bias[112] after B pack in ws

// ---------------------------------------------------------------------------
// Pack weights into bf16 WMMA B-fragments (dense 32x16 KxN, ISA 7.12.2):
// lane = N column (0..15 per half-wave); lane half h holds K=8h..8h+7 (e<8)
// and K=16+8h..16+8h+7 (e>=8), matching A. Fragment (nt,kt) stored as
// 32 lanes x 16 bf16, lane-contiguous (32B per lane).
// ---------------------------------------------------------------------------
__global__ void roi_prepack_kernel(const float* __restrict__ Wc,
                                   const float* __restrict__ bc,
                                   const float* __restrict__ Wt,
                                   const float* __restrict__ bt,
                                   char* __restrict__ ws) {
    int idx = blockIdx.x * 256 + threadIdx.x;
    if (idx < BPACK_ELEMS) {
        int e    = idx & 15;
        int lane = (idx >> 4) & 31;
        int kt   = (idx >> 9) & 31;
        int nt   = idx >> 14;
        int k    = kt * 32 + ((lane >= 16) ? 8 : 0) + (e < 8 ? e : e + 8);
        int col  = nt * 16 + (lane & 15);
        float v = 0.0f;
        if (col < C_CLS)      v = Wc[k * C_CLS + col];
        else if (col < C_TOT) v = Wt[k * C_TF + (col - C_CLS)];
        ((bf16_t*)ws)[idx] = (bf16_t)v;
    } else if (idx < BPACK_ELEMS + 112) {
        int col = idx - BPACK_ELEMS;
        float v = 0.0f;
        if (col < C_CLS)      v = bc[col];
        else if (col < C_TOT) v = bt[col - C_CLS];
        ((float*)(ws + BIAS_OFF))[col] = v;
    }
}

// A-fragment (16x32 bf16, ISA 7.12.2): lane half selects K base b = 0/8;
// lane holds K in [b, b+8) (elems 0..7) and [b+16, b+24) (elems 8..15).
__device__ __forceinline__ v16bf load_a_frag(const float* __restrict__ p) {
    const f32x4 c0 = *(const f32x4*)(p + 0);
    const f32x4 c1 = *(const f32x4*)(p + 4);
    const f32x4 c2 = *(const f32x4*)(p + 16);
    const f32x4 c3 = *(const f32x4*)(p + 20);
    v16bf a;
#pragma unroll
    for (int e = 0; e < 4; ++e) {
        a[e]      = (bf16_t)c0[e];
        a[4 + e]  = (bf16_t)c1[e];
        a[8 + e]  = (bf16_t)c2[e];
        a[12 + e] = (bf16_t)c3[e];
    }
    return a;
}

__device__ __forceinline__ v8f wmma_bf16(v16bf a, v16bf b, v8f c) {
    return __builtin_amdgcn_wmma_f32_16x16x32_bf16(
        false, a, false, b, (short)0, c, false, false);
}

__global__ __launch_bounds__(256, 2)
void roi_head_gemm_kernel(const float* __restrict__ X,
                          const char*  __restrict__ ws,
                          float* __restrict__ out) {
    extern __shared__ __align__(16) char lds[];

    const int lane  = threadIdx.x & 31;
    const int wave  = threadIdx.x >> 5;
    const int wbase = blockIdx.x * 256 + wave * 32;        // 32 rows per wave
    const int kbase = (lane >= 16) ? 8 : 0;
    const float* r0p = X + (size_t)(wbase + (lane & 15)) * D_K + kbase;
    const float* r1p = r0p + (size_t)16 * D_K;

    v8f acc0[NT], acc1[NT];
    const v8f zero = {0.f, 0.f, 0.f, 0.f, 0.f, 0.f, 0.f, 0.f};
#pragma unroll
    for (int t = 0; t < NT; ++t) { acc0[t] = zero; acc1[t] = zero; }

    const f32x4* srcB = (const f32x4*)ws;

    for (int half = 0; half < 2; ++half) {
        __syncthreads();   // all waves done reading previous LDS chunk
        // ---- async stage of this K-half's B fragments into LDS ----------
        // LDS layout: fragment (nt, ktl) at byte (nt*16 + ktl) * 1024
        for (int i = threadIdx.x; i < CHUNK_F4; i += 256) {
            const int nt  = i >> 10;                       // 1024 f4 per nt
            const int off = i & 1023;
            const f32x4* g =
                srcB + (size_t)(nt * KT + half * KT_HALF) * 64 + off;
            const unsigned ldsoff = (unsigned)i * 16u;
            asm volatile("global_load_async_to_lds_b128 %0, %1, off"
                         :: "v"(ldsoff), "v"(g) : "memory");
        }
        asm volatile("s_wait_asynccnt 0x0" ::: "memory");
        __syncthreads();

        // ---- compute 16 K-tiles against the staged half -----------------
#pragma unroll 2
        for (int ktl = 0; ktl < KT_HALF; ++ktl) {
            const int kt = half * KT_HALF + ktl;
            if (kt + 6 < KT) {                             // stream-ahead
                __builtin_prefetch(r0p + (kt + 6) * 32, 0, 3);
                __builtin_prefetch(r1p + (kt + 6) * 32, 0, 3);
            }
            const v16bf a0 = load_a_frag(r0p + kt * 32);
            const v16bf a1 = load_a_frag(r1p + kt * 32);
#pragma unroll
            for (int t = 0; t < NT; ++t) {
                const v16bf b = *(const v16bf*)(
                    lds + (((t * KT_HALF + ktl) * 32 + lane) << 5));
                acc0[t] = wmma_bf16(a0, b, acc0[t]);       // B reused twice
                acc1[t] = wmma_bf16(a1, b, acc1[t]);
            }
        }
    }

    // ---- epilogue: C/D layout lanes0-15 -> M=v, lanes16-31 -> M=v+8 ------
    const float* bias = (const float*)(ws + BIAS_OFF);
    const int rhalf = (lane >= 16) ? 8 : 0;
    float* o_tf = out + (size_t)N_ROWS * C_CLS;
#pragma unroll
    for (int t = 0; t < NT; ++t) {
        const int col = t * 16 + (lane & 15);
        if (col >= C_TOT) continue;                        // padding columns
        const float bv = bias[col];
#pragma unroll
        for (int v = 0; v < 8; ++v) {
            const int r0 = wbase + rhalf + v;
            const int r1 = r0 + 16;
            const float y0 = acc0[t][v] + bv;
            const float y1 = acc1[t][v] + bv;
            if (col < C_CLS) {
                out[(size_t)r0 * C_CLS + col] = y0;
                out[(size_t)r1 * C_CLS + col] = y1;
            } else {
                o_tf[(size_t)r0 * C_TF + (col - C_CLS)] = y0;
                o_tf[(size_t)r1 * C_TF + (col - C_CLS)] = y1;
            }
        }
    }
}

extern "C" void kernel_launch(void* const* d_in, const int* in_sizes, int n_in,
                              void* d_out, int out_size, void* d_ws, size_t ws_size,
                              hipStream_t stream) {
    const float* X  = (const float*)d_in[0];
    const float* Wc = (const float*)d_in[1];
    const float* bc = (const float*)d_in[2];
    const float* Wt = (const float*)d_in[3];
    const float* bt = (const float*)d_in[4];
    char*  ws  = (char*)d_ws;
    float* out = (float*)d_out;

    const int packThreads = BPACK_ELEMS + 112;
    roi_prepack_kernel<<<(packThreads + 255) / 256, 256, 0, stream>>>(
        Wc, bc, Wt, bt, ws);

    roi_head_gemm_kernel<<<N_ROWS / 256, 256, CHUNK_BYTES, stream>>>(
        X, ws, out);
}
